// PAM_Module_3831110828712
// MI455X (gfx1250) — compile-verified, matrix operand order checked by reading, not run on hardware
//
#include <hip/hip_runtime.h>
#include <hip/hip_bf16.h>

// Problem constants (from reference)
#define BB   16
#define CC   512
#define NN   2048
#define QK   64

typedef __attribute__((ext_vector_type(16))) __bf16 v16bf;
typedef __attribute__((ext_vector_type(8)))  __bf16 v8bf;
typedef __attribute__((ext_vector_type(8)))  float  v8f;

// Load a 16x32 bf16 A/B fragment half-pair for this lane.
// khalf = (lane>>4)*8.  Layout per CDNA5 ISA 7.12.2:
//   elems[0..7]  = K khalf .. khalf+7
//   elems[8..15] = K khalf+16 .. khalf+23
__device__ inline v16bf make_frag(const __bf16* p, int khalf) {
  v8bf lo = *(const v8bf*)(p + khalf);
  v8bf hi = *(const v8bf*)(p + khalf + 16);
  v16bf r;
#pragma unroll
  for (int i = 0; i < 8; ++i) { r[i] = lo[i]; r[i + 8] = hi[i]; }
  return r;
}

__device__ inline v8f wmma_bf16(v16bf a, v16bf b, v8f c) {
  return __builtin_amdgcn_wmma_f32_16x16x32_bf16(
      /*neg_a=*/false, a, /*neg_b=*/false, b,
      /*c_mod=*/(short)0, c, /*reuse_a=*/false, /*reuse_b=*/false);
}

// ---------------------------------------------------------------------------
// Pass 0a: x [B,C,N] f32 -> xT [B,N,C] bf16, LDS-tiled 32x32 transpose.
// ---------------------------------------------------------------------------
__global__ __launch_bounds__(256) void convert_x(const float* __restrict__ x,
                                                 __bf16* __restrict__ xT) {
  __shared__ float tile[32][33];
  const int b = blockIdx.z;
  const int n0 = blockIdx.x * 32;
  const int c0 = blockIdx.y * 32;
  const int tx = threadIdx.x & 31;
  const int ty = threadIdx.x >> 5;            // 0..7
#pragma unroll
  for (int i = 0; i < 4; ++i) {
    const int c = ty + 8 * i;
    tile[c][tx] = x[((size_t)b * CC + c0 + c) * NN + n0 + tx];
  }
  __syncthreads();
#pragma unroll
  for (int i = 0; i < 4; ++i) {
    const int n = ty + 8 * i;
    xT[((size_t)b * NN + n0 + n) * CC + c0 + tx] = (__bf16)tile[tx][n];
  }
}

// ---------------------------------------------------------------------------
// Pass 0b: weights f32 -> bf16 (Wq, Wk: [64,512]; Wv: [512,512])
// ---------------------------------------------------------------------------
__global__ __launch_bounds__(256) void convert_w(
    const float* __restrict__ Wq, const float* __restrict__ Wk,
    const float* __restrict__ Wv, __bf16* __restrict__ wq,
    __bf16* __restrict__ wk, __bf16* __restrict__ wv) {
  int i = blockIdx.x * 256 + threadIdx.x;
  const int SQ = QK * CC;                     // 32768
  if (i < SQ)                    wq[i] = (__bf16)Wq[i];
  else if (i < 2 * SQ)           wk[i - SQ] = (__bf16)Wk[i - SQ];
  else if (i < 2 * SQ + CC * CC) wv[i - 2 * SQ] = (__bf16)Wv[i - 2 * SQ];
}

// ---------------------------------------------------------------------------
// Pass 1: q/k GEMM.  out[b,n,o] = sum_c xT[b,n,c]*W[o,c] + bias[o], bf16 out.
// One wave computes a 16(n) x 64(o) strip.  Software-pipelined: fragments of
// chunk k+1 are loaded (independent registers) while chunk k's WMMAs run, so
// the WAR hazard that forced s_wait_loadcnt 0 per WMMA is gone.
// ---------------------------------------------------------------------------
__global__ __launch_bounds__(32) void qk_gemm(const __bf16* __restrict__ xT,
                                              const __bf16* __restrict__ W,
                                              const float* __restrict__ bias,
                                              __bf16* __restrict__ out) {
  const int lane = threadIdx.x & 31;
  const int hi = lane >> 4, lo16 = lane & 15, khalf = hi * 8;
  const int n0 = blockIdx.x * 16, b = blockIdx.y;

  const __bf16* arow = xT + ((size_t)b * NN + n0 + lo16) * CC;
  const __bf16* brow = W + (size_t)lo16 * CC;   // + ot*16*CC per tile

  v8f acc[4];
#pragma unroll
  for (int ot = 0; ot < 4; ++ot) acc[ot] = (v8f){};

  v16bf A  = make_frag(arow, khalf);
  v16bf B0 = make_frag(brow, khalf);
  v16bf B1 = make_frag(brow + 16 * CC, khalf);
  v16bf B2 = make_frag(brow + 32 * CC, khalf);
  v16bf B3 = make_frag(brow + 48 * CC, khalf);
#pragma unroll 2
  for (int kc = 32; kc < CC; kc += 32) {
    v16bf An  = make_frag(arow + kc, khalf);
    v16bf Bn0 = make_frag(brow + kc, khalf);
    v16bf Bn1 = make_frag(brow + 16 * CC + kc, khalf);
    v16bf Bn2 = make_frag(brow + 32 * CC + kc, khalf);
    v16bf Bn3 = make_frag(brow + 48 * CC + kc, khalf);
    acc[0] = wmma_bf16(A, B0, acc[0]);
    acc[1] = wmma_bf16(A, B1, acc[1]);
    acc[2] = wmma_bf16(A, B2, acc[2]);
    acc[3] = wmma_bf16(A, B3, acc[3]);
    A = An; B0 = Bn0; B1 = Bn1; B2 = Bn2; B3 = Bn3;
  }
  acc[0] = wmma_bf16(A, B0, acc[0]);
  acc[1] = wmma_bf16(A, B1, acc[1]);
  acc[2] = wmma_bf16(A, B2, acc[2]);
  acc[3] = wmma_bf16(A, B3, acc[3]);

#pragma unroll
  for (int ot = 0; ot < 4; ++ot) {
    const float bval = bias[ot * 16 + lo16];  // bias depends on column o
#pragma unroll
    for (int r = 0; r < 8; ++r) {
      const int row = r + 8 * hi;             // position n
      out[((size_t)b * NN + n0 + row) * QK + ot * 16 + lo16] =
          (__bf16)(acc[ot][r] + bval);
    }
  }
}

// ---------------------------------------------------------------------------
// Pass 2: v GEMM.  v[b,c,n] = sum_cc Wv[c,cc]*x[b,cc,n] + bv[c], bf16 out.
// 16(c) x 64(n) strip per wave, software-pipelined as above.
// Output stored [B,C,N] so the PV GEMM's B-operand is K(=n)-contiguous.
// ---------------------------------------------------------------------------
__global__ __launch_bounds__(32) void v_gemm(const __bf16* __restrict__ Wv,
                                             const __bf16* __restrict__ xT,
                                             const float* __restrict__ bv,
                                             __bf16* __restrict__ vout) {
  const int lane = threadIdx.x & 31;
  const int hi = lane >> 4, lo16 = lane & 15, khalf = hi * 8;
  const int n0 = blockIdx.x * 64, o0 = blockIdx.y * 16, b = blockIdx.z;

  const __bf16* arow = Wv + (size_t)(o0 + lo16) * CC;
  const __bf16* brow = xT + ((size_t)b * NN + n0 + lo16) * CC;

  v8f acc[4];
#pragma unroll
  for (int nt = 0; nt < 4; ++nt) acc[nt] = (v8f){};

  v16bf A  = make_frag(arow, khalf);
  v16bf B0 = make_frag(brow, khalf);
  v16bf B1 = make_frag(brow + 16 * CC, khalf);
  v16bf B2 = make_frag(brow + 32 * CC, khalf);
  v16bf B3 = make_frag(brow + 48 * CC, khalf);
#pragma unroll 2
  for (int kc = 32; kc < CC; kc += 32) {
    v16bf An  = make_frag(arow + kc, khalf);
    v16bf Bn0 = make_frag(brow + kc, khalf);
    v16bf Bn1 = make_frag(brow + 16 * CC + kc, khalf);
    v16bf Bn2 = make_frag(brow + 32 * CC + kc, khalf);
    v16bf Bn3 = make_frag(brow + 48 * CC + kc, khalf);
    acc[0] = wmma_bf16(A, B0, acc[0]);
    acc[1] = wmma_bf16(A, B1, acc[1]);
    acc[2] = wmma_bf16(A, B2, acc[2]);
    acc[3] = wmma_bf16(A, B3, acc[3]);
    A = An; B0 = Bn0; B1 = Bn1; B2 = Bn2; B3 = Bn3;
  }
  acc[0] = wmma_bf16(A, B0, acc[0]);
  acc[1] = wmma_bf16(A, B1, acc[1]);
  acc[2] = wmma_bf16(A, B2, acc[2]);
  acc[3] = wmma_bf16(A, B3, acc[3]);

#pragma unroll
  for (int r = 0; r < 8; ++r) {
    const int o = o0 + r + 8 * hi;            // channel (row of tile)
    const float bval = bv[o];
#pragma unroll
    for (int nt = 0; nt < 4; ++nt)
      vout[((size_t)b * CC + o) * NN + n0 + nt * 16 + lo16] =
          (__bf16)(acc[nt][r] + bval);
  }
}

// ---------------------------------------------------------------------------
// Pass 3: fused attention.  One block (4 waves) per (batch, 16-query tile).
//  Phase 1: S[16,2048] = q_tile · k^T   (WMMA, strip stays in 128 KB LDS)
//  Phase 2: row softmax; exp packed in-place to bf16 (race-free per-segment)
//  Phase 3: O = P · V^T, software-pipelined at half-chunk granularity
//  Phase 4: out = gamma * O / rowsum + x  (float4 vectorized)
// ---------------------------------------------------------------------------
__global__ __launch_bounds__(128) void attn(const __bf16* __restrict__ qb,
                                            const __bf16* __restrict__ kb,
                                            const __bf16* __restrict__ vb,
                                            const float* __restrict__ x,
                                            const float* __restrict__ gamma_p,
                                            float* __restrict__ out) {
  __shared__ float S[16 * NN];     // 128 KB: energy strip (reused for bf16 P)
  __shared__ float red[16][8];
  __shared__ float rowmax[16];
  __shared__ float rinv[16];

  const int tid = threadIdx.x;
  const int wave = tid >> 5, lane = tid & 31;
  const int hi = lane >> 4, lo16 = lane & 15, khalf = hi * 8;
  const int n0 = blockIdx.x * 16, b = blockIdx.y;

  // ---- Phase 1: energy --------------------------------------------------
  const __bf16* qrow = qb + ((size_t)b * NN + n0 + lo16) * QK;
  v16bf qa0 = make_frag(qrow, khalf);          // K = o 0..31
  v16bf qa1 = make_frag(qrow + 32, khalf);     // K = o 32..63
  for (int jt = wave * 32; jt < wave * 32 + 32; ++jt) {
    const int j0 = jt * 16;
    const __bf16* krow = kb + ((size_t)b * NN + j0 + lo16) * QK;
    v16bf kb0 = make_frag(krow, khalf);
    v16bf kb1 = make_frag(krow + 32, khalf);
    v8f acc = {};
    acc = wmma_bf16(qa0, kb0, acc);
    acc = wmma_bf16(qa1, kb1, acc);
#pragma unroll
    for (int r = 0; r < 8; ++r)
      S[(r + 8 * hi) * NN + j0 + lo16] = acc[r];
  }
  __syncthreads();

  // ---- Phase 2: softmax (16 rows x 8 segments of 256 cols, f4 vector) ----
  {
    const int row = tid >> 3, seg = tid & 7;
    const float4* s4 = (const float4*)(S + row * NN + seg * 256);
    float m = -3.4e38f;
    for (int t = 0; t < 64; ++t) {
      float4 v = s4[t];
      m = fmaxf(m, fmaxf(fmaxf(v.x, v.y), fmaxf(v.z, v.w)));
    }
    red[row][seg] = m;
  }
  __syncthreads();
  if (tid < 16) {
    float m = red[tid][0];
#pragma unroll
    for (int s = 1; s < 8; ++s) m = fmaxf(m, red[tid][s]);
    rowmax[tid] = m;
  }
  __syncthreads();
  {
    const int row = tid >> 3, seg = tid & 7;
    const float m = rowmax[row];
    float* srow = S + row * NN + seg * 256;
    const float4* s4 = (const float4*)srow;
    uint2* p2 = (uint2*)srow;   // in-place pack: writes dwords 2t..2t+1 <= reads 4t..4t+3
    float sum = 0.f;
    for (int t = 0; t < 64; ++t) {
      float4 v = s4[t];
      const float e0 = __expf(v.x - m), e1 = __expf(v.y - m);
      const float e2 = __expf(v.z - m), e3 = __expf(v.w - m);
      sum += (e0 + e1) + (e2 + e3);
      union { __bf16 h[4]; uint2 u; } pk;
      pk.h[0] = (__bf16)e0; pk.h[1] = (__bf16)e1;
      pk.h[2] = (__bf16)e2; pk.h[3] = (__bf16)e3;
      p2[t] = pk.u;
    }
    red[row][seg] = sum;
  }
  __syncthreads();
  if (tid < 16) {
    float s = 0.f;
#pragma unroll
    for (int t = 0; t < 8; ++t) s += red[tid][t];
    rinv[tid] = 1.0f / s;          // defer normalization to the epilogue
  }
  __syncthreads();

  // ---- Phase 3: O = P · V^T ; wave owns channels [wave*128, +128) --------
  // Packed P element (row, K=j): segment s=j>>8, bf16 addr row*4096+s*512+(j&255)
  // Pipelined: B[4..7] loads issue, WMMA 0..3 run; next chunk's A+B[0..3]
  // loads issue, WMMA 4..7 run.  Final prefetch wraps in-bounds (discarded).
  const __bf16* Sbf = (const __bf16*)S;
  const int c0 = wave * 128;
  v8f acc[8];
#pragma unroll
  for (int ct = 0; ct < 8; ++ct) acc[ct] = (v8f){};

  const __bf16* vbase = vb + ((size_t)b * CC + c0 + lo16) * NN;
  const __bf16* prow = Sbf + (size_t)lo16 * 4096;  // this lane's packed P row

  v16bf A = make_frag(prow, khalf);                // jc = 0: seg 0, off 0
  v16bf Ba[4];
#pragma unroll
  for (int ct = 0; ct < 4; ++ct)
    Ba[ct] = make_frag(vbase + (size_t)ct * 16 * NN, khalf);

  for (int jc = 0; jc < NN; jc += 32) {
    v16bf Bb[4];
#pragma unroll
    for (int ct = 0; ct < 4; ++ct)
      Bb[ct] = make_frag(vbase + (size_t)(ct + 4) * 16 * NN + jc, khalf);

    acc[0] = wmma_bf16(A, Ba[0], acc[0]);
    acc[1] = wmma_bf16(A, Ba[1], acc[1]);
    acc[2] = wmma_bf16(A, Ba[2], acc[2]);
    acc[3] = wmma_bf16(A, Ba[3], acc[3]);

    const int jn = (jc + 32) & (NN - 1);           // wrap keeps prefetch in-bounds
    v16bf An = make_frag(prow + (jn >> 8) * 512 + (jn & 255), khalf);
    v16bf Bn[4];
#pragma unroll
    for (int ct = 0; ct < 4; ++ct)
      Bn[ct] = make_frag(vbase + (size_t)ct * 16 * NN + jn, khalf);

    acc[4] = wmma_bf16(A, Bb[0], acc[4]);
    acc[5] = wmma_bf16(A, Bb[1], acc[5]);
    acc[6] = wmma_bf16(A, Bb[2], acc[6]);
    acc[7] = wmma_bf16(A, Bb[3], acc[7]);

    A = An;
#pragma unroll
    for (int ct = 0; ct < 4; ++ct) Ba[ct] = Bn[ct];
  }

  // ---- Phase 4: epilogue: out = gamma*O/rowsum + x (float4) --------------
  const float g = gamma_p[0];
  const float4 r0 = *(const float4*)(rinv + 8 * hi);
  const float4 r1 = *(const float4*)(rinv + 8 * hi + 4);
#pragma unroll
  for (int ct = 0; ct < 8; ++ct) {
    const int ch = c0 + ct * 16 + lo16;
    const float* xr = x + ((size_t)b * CC + ch) * NN + n0 + 8 * hi;
    float* orow = out + ((size_t)b * CC + ch) * NN + n0 + 8 * hi;
    const float4 x0 = ((const float4*)xr)[0];
    const float4 x1 = ((const float4*)xr)[1];
    float4 o0, o1;
    o0.x = g * acc[ct][0] * r0.x + x0.x;
    o0.y = g * acc[ct][1] * r0.y + x0.y;
    o0.z = g * acc[ct][2] * r0.z + x0.z;
    o0.w = g * acc[ct][3] * r0.w + x0.w;
    o1.x = g * acc[ct][4] * r1.x + x1.x;
    o1.y = g * acc[ct][5] * r1.y + x1.y;
    o1.z = g * acc[ct][6] * r1.z + x1.z;
    o1.w = g * acc[ct][7] * r1.w + x1.w;
    ((float4*)orow)[0] = o0;
    ((float4*)orow)[1] = o1;
  }
}

// ---------------------------------------------------------------------------
extern "C" void kernel_launch(void* const* d_in, const int* in_sizes, int n_in,
                              void* d_out, int out_size, void* d_ws,
                              size_t ws_size, hipStream_t stream) {
  (void)in_sizes; (void)n_in; (void)out_size; (void)ws_size;
  const float* x     = (const float*)d_in[0];
  const float* Wq    = (const float*)d_in[1];
  const float* bq    = (const float*)d_in[2];
  const float* Wk    = (const float*)d_in[3];
  const float* bk    = (const float*)d_in[4];
  const float* Wv    = (const float*)d_in[5];
  const float* bv    = (const float*)d_in[6];
  const float* gamma = (const float*)d_in[7];
  float* out = (float*)d_out;

  char* ws = (char*)d_ws;
  __bf16* xT  = (__bf16*)(ws);                    // B*N*C bf16 = 32 MiB
  __bf16* qbf = (__bf16*)(ws + 33554432);         // B*N*64   =  4 MiB
  __bf16* kbf = (__bf16*)(ws + 37748736);         // B*N*64   =  4 MiB
  __bf16* vbf = (__bf16*)(ws + 41943040);         // B*C*N    = 32 MiB
  __bf16* wqb = (__bf16*)(ws + 75497472);         // 64*512   = 64 KiB
  __bf16* wkb = (__bf16*)(ws + 75563008);         // 64*512   = 64 KiB
  __bf16* wvb = (__bf16*)(ws + 75628544);         // 512*512  = 512 KiB

  dim3 gcx(NN / 32, CC / 32, BB);                 // (64, 16, 16)
  convert_x<<<gcx, 256, 0, stream>>>(x, xT);
  convert_w<<<(2 * QK * CC + CC * CC + 255) / 256, 256, 0, stream>>>(
      Wq, Wk, Wv, wqb, wkb, wvb);

  dim3 gqk(NN / 16, BB);                          // (128, 16), 64 o per wave
  qk_gemm<<<gqk, 32, 0, stream>>>(xT, wqb, bq, qbf);
  qk_gemm<<<gqk, 32, 0, stream>>>(xT, wkb, bk, kbf);

  dim3 gv(NN / 64, CC / 16, BB);                  // (32, 32, 16)
  v_gemm<<<gv, 32, 0, stream>>>(wvb, xT, bv, vbf);

  dim3 ga(NN / 16, BB);                           // (128, 16)
  attn<<<ga, 128, 0, stream>>>(qbf, kbf, vbf, x, gamma, out);
}
